// FrequencyFilter_88802743812147
// MI455X (gfx1250) — matrix operand order repaired and checked by reading, not compiled, not run on hardware
//
#include <hip/hip_runtime.h>
#include <hip/hip_bf16.h>
#include <math.h>

// ---------------------------------------------------------------------------
// FrequencyFilter on MI455X (gfx1250, wave32):
//   k0: W (f32) -> Wb (bf16)                               [bandwidth]
//   k1: EMA scan over batch -> combined = [low|high] bf16  [bandwidth, serial B]
//   k2: gate = sigmoid(combined @ Wb^T + b) via V_WMMA_F32_16X16X32_BF16
//       with double-buffered TENSOR_LOAD_TO_LDS (TDM, TENSORcnt) staging,
//       fused epilogue out = gate*low + (1-gate)*high
// ---------------------------------------------------------------------------

typedef __bf16 bf16_t;
typedef __attribute__((ext_vector_type(16))) __bf16 v16bf;
typedef __attribute__((ext_vector_type(8)))  __bf16 v8bf;
typedef __attribute__((ext_vector_type(4)))  __bf16 v4bf;
typedef __attribute__((ext_vector_type(8)))  float  v8f;

typedef unsigned int u32x4 __attribute__((ext_vector_type(4)));
typedef int          i32x4 __attribute__((ext_vector_type(4)));
typedef int          i32x8 __attribute__((ext_vector_type(8)));
typedef __attribute__((address_space(3))) void as3_void;

constexpr int   Bm = 8192;       // batch rows (GEMM M)
constexpr int   Dn = 4096;       // features  (GEMM N)
constexpr int   Kk = 2 * Dn;     // GEMM K = 2D
constexpr float kA = 0.1f;       // EMA alpha

constexpr int BM  = 128;     // block tile M
constexpr int BN  = 128;     // block tile N
constexpr int KC  = 64;      // K-chunk per LDS stage (2 WMMA k-steps)
constexpr int LDT = KC + 8;  // padded LDS row stride (halves): TDM pad fields

__device__ __forceinline__ unsigned lds_byte_off(const void* p) {
  return (unsigned)(unsigned long long)(as3_void*)p;
}

// ---- TDM: one descriptor moves a 128-row x 64-col bf16 tile to LDS --------
// Padded LDS rows: 64 bf16 = 32 DWORDs per row (pad_interval code 4), then
// +4 DWORDs padding (pad_amount code 3) -> row stride 72 bf16 = LDT.
__device__ __forceinline__ void tdm_load_tile(const void* g, void* l,
                                              int tensor_rows) {
  const unsigned long long ga = (unsigned long long)g;

  u32x4 g0;
  g0[0] = 1u;                                    // count=1, user-mode D#
  g0[1] = lds_byte_off(l);                       // lds_addr (bytes)
  g0[2] = (unsigned)ga;                          // global_addr[31:0]
  g0[3] = (unsigned)((ga >> 32) & 0x1FFFFFFu)    // global_addr[56:32]
        | (2u << 30);                            // type = 2 ("image")

  i32x8 g1;
  g1[0] = (int)((1u << 16)                       // data_size = 2 bytes
              | (1u << 20)                       // pad_enable
              | (4u << 22)                       // pad_interval: 32 DWORDs
              | (3u << 25));                     // pad_amount : 4 DWORDs
  g1[1] = (int)(((unsigned)Kk & 0xFFFFu) << 16); // tensor_dim0[15:0] (=8192)
  g1[2] = (int)(((unsigned)tensor_rows & 0xFFFFu) << 16); // tensor_dim1[15:0]
  g1[3] = (int)((unsigned)KC << 16);             // tile_dim0 = 64
  g1[4] = BM;                                    // tile_dim1 = 128, tile_dim2=0
  g1[5] = Kk;                                    // tensor_dim0_stride[31:0]
  g1[6] = 0;                                     // stride[47:32], dim1_stride lo
  g1[7] = 0;

  const i32x4 z4 = {0, 0, 0, 0};                 // groups 2/3 unused (2-D)
  const i32x8 z8 = {0, 0, 0, 0, 0, 0, 0, 0};
#if __has_builtin(__builtin_amdgcn_tensor_load_to_lds)
  // 6-arg form on this toolchain: (g0, g1, g2, g3, g_extra, cpol)
  __builtin_amdgcn_tensor_load_to_lds(g0, g1, z4, z4, z8, 0);
#endif
}

__device__ __forceinline__ void wait_tensor_le2() {
#if __has_builtin(__builtin_amdgcn_s_wait_tensorcnt)
  __builtin_amdgcn_s_wait_tensorcnt((short)2);
#else
  asm volatile("s_wait_tensorcnt 0x2" ::: "memory");
#endif
}
__device__ __forceinline__ void wait_tensor_le0() {
#if __has_builtin(__builtin_amdgcn_s_wait_tensorcnt)
  __builtin_amdgcn_s_wait_tensorcnt((short)0);
#else
  asm volatile("s_wait_tensorcnt 0x0" ::: "memory");
#endif
}

// ------------------------------- kernel 0 ----------------------------------
// W (D x 2D f32) -> Wb (bf16). 8 elements / thread.
__global__ __launch_bounds__(256)
void wconv_kernel(const float* __restrict__ W, bf16_t* __restrict__ Wb) {
  const size_t i = ((size_t)blockIdx.x * 256 + threadIdx.x) * 8;
  const float4* wp = reinterpret_cast<const float4*>(W + i);
  float4 w0 = wp[0], w1 = wp[1];
  v8bf v;
  v[0] = (bf16_t)w0.x; v[1] = (bf16_t)w0.y;
  v[2] = (bf16_t)w0.z; v[3] = (bf16_t)w0.w;
  v[4] = (bf16_t)w1.x; v[5] = (bf16_t)w1.y;
  v[6] = (bf16_t)w1.z; v[7] = (bf16_t)w1.w;
  *reinterpret_cast<v8bf*>(Wb + i) = v;
}

// ------------------------------- kernel 1 ----------------------------------
// One thread owns 4 consecutive columns; serial scan over all 8192 rows.
__global__ __launch_bounds__(256)
void ema_scan_kernel(const float* __restrict__ z, bf16_t* __restrict__ comb) {
  const int col4 = (blockIdx.x * blockDim.x + threadIdx.x) << 2;
  const float4* zp = reinterpret_cast<const float4*>(z) + (col4 >> 2);
  const int zstride4 = Dn >> 2;

  float4 zi  = zp[0];     // row 0: ema = z0, high = 0
  float4 ema = zi;

  for (int i = 0; i < Bm; ++i) {
    if (i > 0) {
      zi = zp[(size_t)i * zstride4];
      ema.x = fmaf(kA, zi.x - ema.x, ema.x);
      ema.y = fmaf(kA, zi.y - ema.y, ema.y);
      ema.z = fmaf(kA, zi.z - ema.z, ema.z);
      ema.w = fmaf(kA, zi.w - ema.w, ema.w);
    }
    v4bf lo, hi;
    lo[0] = (bf16_t)ema.x;  hi[0] = (bf16_t)(zi.x - ema.x);
    lo[1] = (bf16_t)ema.y;  hi[1] = (bf16_t)(zi.y - ema.y);
    lo[2] = (bf16_t)ema.z;  hi[2] = (bf16_t)(zi.z - ema.z);
    lo[3] = (bf16_t)ema.w;  hi[3] = (bf16_t)(zi.w - ema.w);
    *reinterpret_cast<v4bf*>(comb + (size_t)i * Kk + col4)      = lo;
    *reinterpret_cast<v4bf*>(comb + (size_t)i * Kk + Dn + col4) = hi;
  }
}

// ------------------------------- kernel 2 ----------------------------------
__global__ __launch_bounds__(256)
void gate_gemm_kernel(const bf16_t* __restrict__ comb,
                      const bf16_t* __restrict__ Wb,
                      const float*  __restrict__ bias,
                      float*        __restrict__ out) {
  __shared__ __align__(16) bf16_t sA[2][BM * LDT];  // combined tiles (dbl buf)
  __shared__ __align__(16) bf16_t sB[2][BN * LDT];  // Wb tiles      (dbl buf)

  const int tid  = threadIdx.x;
  const int lane = tid & 31;
  const int wave = tid >> 5;       // 8 waves
  const int wm   = wave & 3;       // 4 waves along M: 32 rows each
  const int wn   = wave >> 2;      // 2 waves along N: 64 cols each
  const int blockM = blockIdx.y * BM;
  const int blockN = blockIdx.x * BN;

  const int r  = lane & 15;
  const int hi = lane >> 4;

  // One TDM stage = 2 descriptors (A tile + B tile), issued by wave 0 only.
  auto issue_stage = [&](int buf, int k0) {
    tdm_load_tile(&comb[(size_t)blockM * Kk + k0], &sA[buf][0], Bm);
    tdm_load_tile(&Wb[(size_t)blockN * Kk + k0],   &sB[buf][0], Dn);
  };

  v8f acc[2][4];
  const v8f vzero = {};
#pragma unroll
  for (int tm = 0; tm < 2; ++tm)
#pragma unroll
    for (int tn = 0; tn < 4; ++tn) acc[tm][tn] = vzero;

  if (wave == 0) issue_stage(0, 0);  // prologue

  for (int k0 = 0; k0 < Kk; k0 += KC) {
    const int  cur      = (k0 / KC) & 1;
    const bool has_next = (k0 + KC) < Kk;

    if (wave == 0) {
      if (has_next) {
        issue_stage(cur ^ 1, k0 + KC);  // DMA overlaps compute below
        wait_tensor_le2();              // cur stage (2 older DMAs) landed
      } else {
        wait_tensor_le0();
      }
    }
    __syncthreads();                    // cur tiles visible to all waves

    // ---- 2 WMMA k-steps on the staged chunk -----------------------------
#pragma unroll
    for (int kk = 0; kk < KC; kk += 32) {
      // A fragments (16x32 bf16): lanes 0-15 K 0-7 & 16-23 of row M=lane;
      // lanes 16-31 K 8-15 & 24-31 (ISA 7.12.2).
      v16bf afrag[2];
#pragma unroll
      for (int tm = 0; tm < 2; ++tm) {
        const int row = wm * 32 + tm * 16 + r;
        const int kb  = kk + hi * 8;
        v8bf a0 = *reinterpret_cast<const v8bf*>(&sA[cur][row * LDT + kb]);
        v8bf a1 = *reinterpret_cast<const v8bf*>(&sA[cur][row * LDT + kb + 16]);
        afrag[tm] = __builtin_shufflevector(a0, a1, 0, 1, 2, 3, 4, 5, 6, 7,
                                            8, 9, 10, 11, 12, 13, 14, 15);
      }
      // B fragments (32x16 bf16): lane n<16 -> col n, K 0-7 & 16-23; lane>=16 ->
      // col n-16, K 8-15 & 24-31. Wb rows are K-contiguous.
      v16bf bfrag[4];
#pragma unroll
      for (int tn = 0; tn < 4; ++tn) {
        const int n  = wn * 64 + tn * 16 + r;
        const int kb = kk + hi * 8;
        v8bf b0 = *reinterpret_cast<const v8bf*>(&sB[cur][n * LDT + kb]);
        v8bf b1 = *reinterpret_cast<const v8bf*>(&sB[cur][n * LDT + kb + 16]);
        bfrag[tn] = __builtin_shufflevector(b0, b1, 0, 1, 2, 3, 4, 5, 6, 7,
                                            8, 9, 10, 11, 12, 13, 14, 15);
      }
#pragma unroll
      for (int tm = 0; tm < 2; ++tm)
#pragma unroll
        for (int tn = 0; tn < 4; ++tn)
          acc[tm][tn] = __builtin_amdgcn_wmma_f32_16x16x32_bf16(
              false, afrag[tm], false, bfrag[tn],
              (short)0, acc[tm][tn], false, false);
    }
    __syncthreads();  // all reads of cur done before it is re-issued
  }

  // ---- epilogue: gate = sigmoid(acc + b); out = g*low + (1-g)*high --------
#pragma unroll
  for (int tm = 0; tm < 2; ++tm) {
#pragma unroll
    for (int tn = 0; tn < 4; ++tn) {
      const int ncol = blockN + wn * 64 + tn * 16 + r;
      const float bb = bias[ncol];
#pragma unroll
      for (int e = 0; e < 8; ++e) {
        const int mrow = blockM + wm * 32 + tm * 16 + e + hi * 8;
        const float x  = acc[tm][tn][e] + bb;
        const float g  = 1.0f / (1.0f + __expf(-x));
        const float lo = (float)comb[(size_t)mrow * Kk + ncol];
        const float hh = (float)comb[(size_t)mrow * Kk + Dn + ncol];
        out[(size_t)mrow * Dn + ncol] = g * lo + (1.0f - g) * hh;
      }
    }
  }
}

// ------------------------------- launcher ----------------------------------
extern "C" void kernel_launch(void* const* d_in, const int* in_sizes, int n_in,
                              void* d_out, int out_size, void* d_ws, size_t ws_size,
                              hipStream_t stream) {
  const float* z  = (const float*)d_in[0];   // (B, D) f32
  const float* W  = (const float*)d_in[1];   // (D, 2D) f32
  const float* b  = (const float*)d_in[2];   // (D,)   f32
  float* out      = (float*)d_out;           // (B, D) f32

  // workspace: comb (B x 2D bf16, 128 MiB) then Wb (D x 2D bf16, 64 MiB)
  bf16_t* comb = (bf16_t*)d_ws;
  bf16_t* Wb   = (bf16_t*)((char*)d_ws + (size_t)Bm * Kk * sizeof(bf16_t));

  wconv_kernel<<<dim3((size_t)Dn * Kk / 8 / 256), 256, 0, stream>>>(W, Wb);
  ema_scan_kernel<<<dim3(Dn / (256 * 4)), 256, 0, stream>>>(z, comb);
  gate_gemm_kernel<<<dim3(Dn / BN, Bm / BM), 256, 0, stream>>>(comb, Wb, b, out);
}
